// MultiScaleTemporalAttention_83270825935089
// MI455X (gfx1250) — compile-verified, matrix operand order checked by reading, not run on hardware
//
#include <hip/hip_runtime.h>
#include <hip/hip_bf16.h>

// ---------------------------------------------------------------------------
// MultiScaleTemporalAttention for MI455X (gfx1250, wave32, WMMA)
// B=8, T=2048, C=1024, H=128.  All matmuls via v_wmma_f32_16x16x32_bf16.
// ---------------------------------------------------------------------------

#define BSZ 8
#define TLEN 2048
#define CDIM 1024
#define HDIM 128
#define BT (BSZ * TLEN)

typedef __bf16 v2bf  __attribute__((ext_vector_type(2)));
typedef __bf16 v16bf __attribute__((ext_vector_type(16)));
typedef float  v8f   __attribute__((ext_vector_type(8)));
// reduced-alignment view for LDS rows padded to 40 halves (80B, 16B-aligned)
typedef v16bf v16bf_a16 __attribute__((aligned(16)));

// workspace layout (bytes)
#define OFF_WT   0                            // 3 * H * C bf16  = 786432
#define OFF_Q    (OFF_WT + 3 * HDIM * CDIM * 2)
#define OFF_K    (OFF_Q  + BT * HDIM * 2)     // 4 MB each
#define OFF_V    (OFF_K  + BT * HDIM * 2)     // V stored transposed (B,H,T)
#define OFF_BIAS (OFF_V  + BT * HDIM * 2)     // T floats

// ---------------------------------------------------------------------------
// Kernel 1: W (C x H, f32) -> W^T (H x C, bf16) for q,k,v
// ---------------------------------------------------------------------------
__global__ void wt_convert_kernel(const float* __restrict__ Wq,
                                  const float* __restrict__ Wk,
                                  const float* __restrict__ Wv,
                                  __bf16* __restrict__ WT) {
  int idx = blockIdx.x * blockDim.x + threadIdx.x;
  if (idx >= 3 * CDIM * HDIM) return;
  int mat = idx / (CDIM * HDIM);
  int rem = idx - mat * (CDIM * HDIM);
  int c = rem / HDIM, h = rem % HDIM;
  const float* W = (mat == 0) ? Wq : ((mat == 1) ? Wk : Wv);
  WT[(size_t)mat * HDIM * CDIM + (size_t)h * CDIM + c] = (__bf16)W[(size_t)c * HDIM + h];
}

// ---------------------------------------------------------------------------
// Kernel 2: temporal bias table, bias[d] = log(sum_i alpha_i * g_i^d + 1e-8)
// ---------------------------------------------------------------------------
__global__ void bias_table_kernel(const float* __restrict__ gfl,
                                  const float* __restrict__ gml,
                                  const float* __restrict__ gsl,
                                  const float* __restrict__ al,
                                  float* __restrict__ biasTab) {
  int d = blockIdx.x * blockDim.x + threadIdx.x;
  if (d >= TLEN) return;
  float gf = 0.3f  / (1.f + __expf(-gfl[0])) + 0.6f;
  float gm = 0.15f / (1.f + __expf(-gml[0])) + 0.85f;
  float gs = 0.05f / (1.f + __expf(-gsl[0])) + 0.95f;
  float a0 = al[0], a1 = al[1], a2 = al[2];
  float amax = fmaxf(a0, fmaxf(a1, a2));
  float e0 = __expf(a0 - amax), e1 = __expf(a1 - amax), e2 = __expf(a2 - amax);
  float inv = 1.f / (e0 + e1 + e2);
  float fd = (float)d;
  float mix = e0 * inv * __expf(fd * __logf(gf))
            + e1 * inv * __expf(fd * __logf(gm))
            + e2 * inv * __expf(fd * __logf(gs));
  biasTab[d] = __logf(mix + 1e-8f);
}

// ---------------------------------------------------------------------------
// Kernel 3: fused QKV projection.
// One block = 16 output rows; 24 waves = {Q,K,V} x 8 column tiles of 16.
// A tile (16x32 bf16) double-buffered in LDS with a depth-2 software
// pipeline: the global float2 load for chunk k+1 is issued while chunk k's
// WMMA runs; its convert+ds_store happens at the top of iteration k+1, a full
// chunk of latency later.  V is stored transposed (B,H,T) for the PV WMMA.
// ---------------------------------------------------------------------------
__global__ void qkv_gemm_kernel(const float* __restrict__ x,
                                const __bf16* __restrict__ WT,
                                __bf16* __restrict__ Qb,
                                __bf16* __restrict__ Kb,
                                __bf16* __restrict__ Vt) {
  // 40 halves per row: 80B stride -> conflict-free ds_load_b128 across 16 lanes
  __shared__ alignas(32) __bf16 As[2][16][40];

  const int tid  = threadIdx.x;
  const int wave = tid >> 5;
  const int lane = tid & 31;
  const int ln   = lane & 15;
  const int lh   = lane >> 4;
  const int mat     = wave >> 3;      // 0=Q 1=K 2=V
  const int colTile = wave & 7;
  const int rowBase = blockIdx.x * 16;
  const int col     = colTile * 16 + ln;

  const __bf16* wtm = WT + (size_t)mat * HDIM * CDIM;

  // 256 staging threads, one float2 (-> packed 2xbf16) each per 16x32 chunk
  const int sr  = tid >> 4;           // staging row
  const int scp = (tid & 15) * 2;     // staging column pair

  float2 fstage = {0.f, 0.f};
  auto loadStage = [&](int k0) {
    if (tid < 256)
      fstage = *(const float2*)&x[(size_t)(rowBase + sr) * CDIM + k0 + scp];
  };
  auto storeStage = [&](__bf16 (*dst)[40]) {
    if (tid < 256) {
      v2bf p = { (__bf16)fstage.x, (__bf16)fstage.y };
      *(v2bf*)&dst[sr][scp] = p;
    }
  };

  v8f acc = {};
  loadStage(0);
  int buf = 0;
  for (int k0 = 0; k0 < CDIM; k0 += 32) {
    storeStage(As[buf]);                   // data loaded one iteration ago
    __syncthreads();                       // chunk k0 visible in As[buf]
    // A: lane holds row (lane%16), halves k0 + (lane/16)*16 .. +15
    v16bf a = *(const v16bf_a16*)&As[buf][ln][lh * 16];
    // B: lane holds output column `col`, same contiguous k-slice of W^T
    v16bf b = *(const v16bf*)&wtm[(size_t)col * CDIM + k0 + lh * 16];
    if (k0 + 32 < CDIM) loadStage(k0 + 32);  // prefetch overlaps the WMMA
    acc = __builtin_amdgcn_wmma_f32_16x16x32_bf16(false, a, false, b,
                                                  (short)0, acc, false, false);
    buf ^= 1;
  }

#pragma unroll
  for (int r = 0; r < 8; ++r) {            // C/D layout: row = r + 8*(lane/16)
    int grow = rowBase + r + 8 * lh;
    __bf16 val = (__bf16)acc[r];
    if (mat == 0) {
      Qb[(size_t)grow * HDIM + col] = val;
    } else if (mat == 1) {
      Kb[(size_t)grow * HDIM + col] = val;
    } else {
      int bb = grow >> 11;                 // / TLEN
      int t  = grow & (TLEN - 1);
      Vt[((size_t)bb * HDIM + col) * TLEN + t] = val;
    }
  }
}

// ---------------------------------------------------------------------------
// Kernel 4: causal flash attention with temporal bias.
// One wave per (batch, 16-query tile); 32 keys per step (two S tiles).
// ---------------------------------------------------------------------------
__global__ void attn_kernel(const __bf16* __restrict__ Qb,
                            const __bf16* __restrict__ Kb,
                            const __bf16* __restrict__ Vt,
                            const float* __restrict__ biasTab,
                            float* __restrict__ out) {
  __shared__ alignas(32) __bf16 Ps[16][40];   // P tile: 16 queries x 32 keys

  const int lane = threadIdx.x;
  const int ln   = lane & 15;
  const int lh   = lane >> 4;
  const int b    = blockIdx.y;
  const int ti0  = blockIdx.x * 16;
  const float sc = 0.08838834764831845f;      // 1/sqrt(H)

  // Q A-tiles for all 4 k-chunks of H=128 (resident for the whole tile)
  v16bf qa[4];
#pragma unroll
  for (int c = 0; c < 4; ++c)
    qa[c] = *(const v16bf*)&Qb[((size_t)(b * TLEN + ti0 + ln)) * HDIM + c * 32 + lh * 16];

  v8f acc[8] = {};                            // 16 x 128 f32 accumulator
  float mrow[8], lrow[8];
#pragma unroll
  for (int r = 0; r < 8; ++r) { mrow[r] = -INFINITY; lrow[r] = 0.f; }

  for (int jb = 0; jb < ti0 + 16; jb += 32) {
    // hint the next iteration's K rows into cache (global_prefetch path);
    // each 256B row spans two 128B lines
    int jn = jb + 32;
    if (jn < ti0 + 16) {
      const __bf16* pk0 = &Kb[((size_t)(b * TLEN + jn + ln)) * HDIM];
      const __bf16* pk1 = &Kb[((size_t)(b * TLEN + jn + 16 + ln)) * HDIM];
      __builtin_prefetch(pk0, 0, 3);
      __builtin_prefetch(pk0 + 64, 0, 3);
      __builtin_prefetch(pk1, 0, 3);
      __builtin_prefetch(pk1 + 64, 0, 3);
    }

    // ---- S = Q @ K^T for 32 keys (two 16x16 tiles), f32 accumulate ----
    v8f s0 = {}, s1 = {};
#pragma unroll
    for (int c = 0; c < 4; ++c) {
      v16bf k0 = *(const v16bf*)&Kb[((size_t)(b * TLEN + jb + ln)) * HDIM + c * 32 + lh * 16];
      v16bf k1 = *(const v16bf*)&Kb[((size_t)(b * TLEN + jb + 16 + ln)) * HDIM + c * 32 + lh * 16];
      s0 = __builtin_amdgcn_wmma_f32_16x16x32_bf16(false, qa[c], false, k0, (short)0, s0, false, false);
      s1 = __builtin_amdgcn_wmma_f32_16x16x32_bf16(false, qa[c], false, k1, (short)0, s1, false, false);
    }

    // ---- scale + temporal bias + causal mask + online softmax update ----
#pragma unroll
    for (int r = 0; r < 8; ++r) {
      int ti = ti0 + r + 8 * lh;              // query index of this element
      int d0 = ti - (jb + ln);                // delta for key tile 0
      int d1 = d0 - 16;                       // delta for key tile 1
      float b0 = biasTab[d0 >= 0 ? d0 : 0];
      float b1 = biasTab[d1 >= 0 ? d1 : 0];
      float v0 = (d0 >= 0) ? (s0[r] * sc + b0) : -1e30f;
      float v1 = (d1 >= 0) ? (s1[r] * sc + b1) : -1e30f;

      // row-max across the 16 lanes of each half (masks 1..8 stay in-half)
      float mx = fmaxf(v0, v1);
#pragma unroll
      for (int off = 1; off < 16; off <<= 1)
        mx = fmaxf(mx, __shfl_xor(mx, off, 32));

      float mnew = fmaxf(mrow[r], mx);
      float cf = __expf(mrow[r] - mnew);
      float p0 = __expf(v0 - mnew);
      float p1 = __expf(v1 - mnew);

      float rs = p0 + p1;
#pragma unroll
      for (int off = 1; off < 16; off <<= 1)
        rs += __shfl_xor(rs, off, 32);

      lrow[r] = lrow[r] * cf + rs;
      mrow[r] = mnew;
#pragma unroll
      for (int h = 0; h < 8; ++h) acc[h][r] *= cf;

      int m = r + 8 * lh;                     // C-layout -> LDS (row, key)
      Ps[m][ln]      = (__bf16)p0;
      Ps[m][ln + 16] = (__bf16)p1;
    }
    __syncthreads();

    // re-load P in A-matrix layout (row = lane%16, 16 contiguous keys/half)
    v16bf pa = *(const v16bf_a16*)&Ps[ln][lh * 16];

    // ---- O += P @ V  (V pre-transposed: lane = h column, contiguous keys)
#pragma unroll
    for (int h = 0; h < 8; ++h) {
      v16bf vb = *(const v16bf*)&Vt[((size_t)b * HDIM + h * 16 + ln) * TLEN + jb + lh * 16];
      acc[h] = __builtin_amdgcn_wmma_f32_16x16x32_bf16(false, pa, false, vb,
                                                       (short)0, acc[h], false, false);
    }
    __syncthreads();
  }

  // ---- normalize (8 reciprocals instead of 64 divides) and store f32 ----
  float inv[8];
#pragma unroll
  for (int r = 0; r < 8; ++r) inv[r] = 1.0f / lrow[r];
#pragma unroll
  for (int h = 0; h < 8; ++h) {
#pragma unroll
    for (int r = 0; r < 8; ++r) {
      int row = ti0 + r + 8 * lh;
      out[((size_t)b * TLEN + row) * HDIM + h * 16 + ln] = acc[h][r] * inv[r];
    }
  }
}

// ---------------------------------------------------------------------------
extern "C" void kernel_launch(void* const* d_in, const int* in_sizes, int n_in,
                              void* d_out, int out_size, void* d_ws, size_t ws_size,
                              hipStream_t stream) {
  (void)in_sizes; (void)n_in; (void)out_size; (void)ws_size;

  const float* x   = (const float*)d_in[0];
  const float* Wq  = (const float*)d_in[1];
  const float* Wk  = (const float*)d_in[2];
  const float* Wv  = (const float*)d_in[3];
  const float* gfl = (const float*)d_in[4];
  const float* gml = (const float*)d_in[5];
  const float* gsl = (const float*)d_in[6];
  const float* al  = (const float*)d_in[7];

  char* ws = (char*)d_ws;
  __bf16* WT      = (__bf16*)(ws + OFF_WT);
  __bf16* Qb      = (__bf16*)(ws + OFF_Q);
  __bf16* Kb      = (__bf16*)(ws + OFF_K);
  __bf16* Vt      = (__bf16*)(ws + OFF_V);
  float*  biasTab = (float*)(ws + OFF_BIAS);

  wt_convert_kernel<<<(3 * CDIM * HDIM + 255) / 256, 256, 0, stream>>>(Wq, Wk, Wv, WT);
  bias_table_kernel<<<(TLEN + 255) / 256, 256, 0, stream>>>(gfl, gml, gsl, al, biasTab);
  qkv_gemm_kernel<<<BT / 16, 768, 0, stream>>>(x, WT, Qb, Kb, Vt);
  attn_kernel<<<dim3(TLEN / 16, BSZ), 32, 0, stream>>>(Qb, Kb, Vt, biasTab, (float*)d_out);
}
